// Short_term_atention_9835475107903
// MI455X (gfx1250) — compile-verified, matrix operand order checked by reading, not run on hardware
//
#include <hip/hip_runtime.h>
#include <stdint.h>

typedef __attribute__((ext_vector_type(16))) __bf16 v16bf;
typedef __attribute__((ext_vector_type(8)))  float  v8f;
typedef __attribute__((ext_vector_type(2)))  float  f32x2;
typedef __attribute__((ext_vector_type(2)))  __bf16 bf16x2;

union BF16x16 { v16bf v; uint32_t u[8]; };

#define DD 1024
#define SS 2048
#define BB 16

// Native packed f32->bf16 (lowers to v_cvt_pk_bf16_f32 on gfx1250)
__device__ __forceinline__ uint32_t pack_bf16(float a, float b) {
  f32x2 f; f[0] = a; f[1] = b;
  union { bf16x2 h; uint32_t u; } cv;
  cv.h = __builtin_convertvector(f, bf16x2);
  return cv.u;
}
__device__ __forceinline__ uint16_t bf16bits(float a) {
  union { __bf16 h; uint16_t u; } cv;
  cv.h = (__bf16)a;
  return cv.u;
}

// K-index base for VGPR v within a 16-bit A/B fragment (16x32 / 32x16 layout):
// lanes 0-15: VGPR0-3 -> K 0..7 (pairs), VGPR4-7 -> K 16..23
// lanes16-31: VGPR0-3 -> K 8..15,       VGPR4-7 -> K 24..31
__device__ __forceinline__ int kbase(int v, int half) {
  return (v < 4 ? 2 * v : 2 * (v - 4) + 16) + 8 * half;
}

// ---------------------------------------------------------------------------
// Kernel 1: C[M,N] = X[M,K] @ W[K,N], f32 in, bf16 out (scale folded in).
// M = B*S = 32768, N = K = 1024. 128x128 tile per workgroup, 8 waves.
// pair_out=0: row-major bf16.  pair_out=1: key-pair-interleaved layout
//   element (row,col) -> Out32[(row>>1)*D + col] half (row&1), so that the
//   attention P@V B-fragments become single aligned 32-bit loads.
// ---------------------------------------------------------------------------
__global__ __launch_bounds__(256) void proj_gemm(
    const float* __restrict__ X, const float* __restrict__ W,
    uint16_t* __restrict__ Out, float scale, int pair_out) {
  __shared__ uint16_t As[128][32];   // [m][k] bf16
  __shared__ uint16_t Bs[128][32];   // [n][k] bf16 (transposed: pairs contiguous)

  const int tid  = threadIdx.x;
  const int wave = tid >> 5, lane = tid & 31;
  const int half = lane >> 4, l16 = lane & 15;
  const int m0 = blockIdx.x * 128;
  const int n0 = blockIdx.y * 128;
  const int wm = wave & 3;    // 4 waves over M -> 2 M-tiles each
  const int wn = wave >> 2;   // 2 waves over N -> 4 N-tiles each

  v8f acc[2][4];
  for (int i = 0; i < 2; ++i)
    for (int j = 0; j < 4; ++j)
      for (int e = 0; e < 8; ++e) acc[i][j][e] = 0.0f;

  for (int k0 = 0; k0 < DD; k0 += 32) {
    // stage A: thread t -> row t/2, 16 consecutive k; packed 32-bit LDS stores
    {
      const int m  = tid >> 1;
      const int kk = (tid & 1) * 16;
      const float* src = X + (size_t)(m0 + m) * DD + k0 + kk;
#pragma unroll
      for (int i = 0; i < 8; ++i)
        *(uint32_t*)&As[m][kk + 2 * i] = pack_bf16(src[2 * i], src[2 * i + 1]);
    }
    // stage B transposed: thread t -> k row t/8, 16 consecutive n
    {
      const int kb = tid >> 3;
      const int nn = (tid & 7) * 16;
      const float* src = W + (size_t)(k0 + kb) * DD + n0 + nn;
#pragma unroll
      for (int i = 0; i < 16; ++i) Bs[nn + i][kb] = bf16bits(src[i]);
    }
    __syncthreads();

    BF16x16 a[2], bm[4];
#pragma unroll
    for (int i = 0; i < 2; ++i) {
      const int row = (2 * wm + i) * 16 + l16;
#pragma unroll
      for (int v = 0; v < 8; ++v)
        a[i].u[v] = *(const uint32_t*)&As[row][kbase(v, half)];
    }
#pragma unroll
    for (int j = 0; j < 4; ++j) {
      const int col = (4 * wn + j) * 16 + l16;
#pragma unroll
      for (int v = 0; v < 8; ++v)
        bm[j].u[v] = *(const uint32_t*)&Bs[col][kbase(v, half)];
    }
#pragma unroll
    for (int i = 0; i < 2; ++i)
#pragma unroll
      for (int j = 0; j < 4; ++j)
        acc[i][j] = __builtin_amdgcn_wmma_f32_16x16x32_bf16(
            false, a[i].v, false, bm[j].v, (short)0, acc[i][j], false, false);
    __syncthreads();
  }

  // C layout: lane holds N = l16, rows v + 8*half (8 consecutive rows/lane)
  if (pair_out) {
    uint32_t* Out32 = (uint32_t*)Out;
#pragma unroll
    for (int i = 0; i < 2; ++i)
#pragma unroll
      for (int j = 0; j < 4; ++j) {
        const int col = n0 + (4 * wn + j) * 16 + l16;
#pragma unroll
        for (int v = 0; v < 8; v += 2) {
          const int row = m0 + (2 * wm + i) * 16 + v + 8 * half;  // even
          Out32[(size_t)(row >> 1) * DD + col] =
              pack_bf16(acc[i][j][v] * scale, acc[i][j][v + 1] * scale);
        }
      }
  } else {
#pragma unroll
    for (int i = 0; i < 2; ++i)
#pragma unroll
      for (int j = 0; j < 4; ++j) {
        const int col = n0 + (4 * wn + j) * 16 + l16;
#pragma unroll
        for (int v = 0; v < 8; ++v) {
          const int row = m0 + (2 * wm + i) * 16 + v + 8 * half;
          Out[(size_t)row * DD + col] = bf16bits(acc[i][j][v] * scale);
        }
      }
  }
}

// ---------------------------------------------------------------------------
// Kernel 2: causal attention (online softmax) + residual + LayerNorm.
// One 16-query tile per workgroup; 8 waves each own a 128-wide D slice.
// ---------------------------------------------------------------------------
__global__ __launch_bounds__(256) void attn_ln(
    const uint16_t* __restrict__ Qp, const uint16_t* __restrict__ Kp,
    const uint32_t* __restrict__ VpP,   // key-pair-interleaved V'
    const float* __restrict__ Vres,
    const uint8_t* __restrict__ mask, const float* __restrict__ gamma,
    const float* __restrict__ beta, float* __restrict__ Out) {
  __shared__ float    sred[8 * 512];   // per-wave partial score tiles (16 KB)
  __shared__ uint16_t pbuf[16][32];    // softmaxed P tile, bf16, row-major
  __shared__ float    lnsum[8][16];
  __shared__ float    lnsq[8][16];

  const int tid  = threadIdx.x;
  const int wave = tid >> 5, lane = tid & 31;
  const int half = lane >> 4, l16 = lane & 15;
  const int b  = blockIdx.x >> 7;          // 128 query tiles per batch
  const int qt = blockIdx.x & 127;
  const int q0 = qt * 16;
  const size_t rowB = (size_t)b * SS;      // row offset of this batch

  // padding mask per accumulator row (row r = v + 8*half)
  bool rowmask[8];
#pragma unroll
  for (int v = 0; v < 8; ++v)
    rowmask[v] = mask[rowB + q0 + v + 8 * half] != 0;

  // Q fragments over this wave's 128-wide D slice (4 sub-steps of K=32)
  BF16x16 aq[4];
  {
    const uint16_t* qrow = Qp + (rowB + q0 + l16) * DD;
#pragma unroll
    for (int s = 0; s < 4; ++s)
#pragma unroll
      for (int v = 0; v < 8; ++v)
        aq[s].u[v] = *(const uint32_t*)(qrow + wave * 128 + s * 32 + kbase(v, half));
  }

  v8f acc[8];
  float mrow[8], lrow[8];
#pragma unroll
  for (int t = 0; t < 8; ++t)
    for (int e = 0; e < 8; ++e) acc[t][e] = 0.0f;
#pragma unroll
  for (int v = 0; v < 8; ++v) { mrow[v] = -3.0e38f; lrow[v] = 0.0f; }

  const int nsteps = ((q0 + 15) >> 5) + 1;   // causal: keys <= q0+15

  for (int kt = 0; kt < nsteps; ++kt) {
    const int keybase = kt * 32;

    // ---- partial scores over this wave's D slice (K' is L2-resident) ----
    v8f c0, c1;
    for (int e = 0; e < 8; ++e) { c0[e] = 0.0f; c1[e] = 0.0f; }
    const uint16_t* kr0 = Kp + (rowB + keybase + l16) * DD;
    const uint16_t* kr1 = kr0 + (size_t)16 * DD;
    if (kt + 1 < nsteps)
      __builtin_prefetch(kr0 + (size_t)32 * DD + wave * 128, 0, 0);
#pragma unroll
    for (int s = 0; s < 4; ++s) {
      BF16x16 b0, b1;
#pragma unroll
      for (int v = 0; v < 8; ++v) {
        const int d = wave * 128 + s * 32 + kbase(v, half);
        b0.u[v] = *(const uint32_t*)(kr0 + d);
        b1.u[v] = *(const uint32_t*)(kr1 + d);
      }
      c0 = __builtin_amdgcn_wmma_f32_16x16x32_bf16(false, aq[s].v, false, b0.v,
                                                   (short)0, c0, false, false);
      c1 = __builtin_amdgcn_wmma_f32_16x16x32_bf16(false, aq[s].v, false, b1.v,
                                                   (short)0, c1, false, false);
    }

    // ---- cross-wave reduction of the 16x32 score tile ----
    {
      float* my = sred + wave * 512;
#pragma unroll
      for (int v = 0; v < 8; ++v) {
        const int off = (v + 8 * half) * 16 + l16;
        my[off]       = c0[v];
        my[256 + off] = c1[v];
      }
    }
    __syncthreads();
    float s0[8], s1[8];
#pragma unroll
    for (int v = 0; v < 8; ++v) {
      const int off = (v + 8 * half) * 16 + l16;
      float t0 = 0.0f, t1 = 0.0f;
#pragma unroll
      for (int w = 0; w < 8; ++w) {
        t0 += sred[w * 512 + off];
        t1 += sred[w * 512 + 256 + off];
      }
      s0[v] = t0; s1[v] = t1;
    }
    __syncthreads();

    // ---- mask + online softmax (rows live across 16-lane halves) ----
#pragma unroll
    for (int v = 0; v < 8; ++v) {
      const int r = q0 + v + 8 * half;
      if (rowmask[v] || keybase + l16 > r)      s0[v] = -1.0e10f;
      if (rowmask[v] || keybase + 16 + l16 > r) s1[v] = -1.0e10f;
      float tmax = fmaxf(s0[v], s1[v]);
      for (int o = 1; o < 16; o <<= 1) tmax = fmaxf(tmax, __shfl_xor(tmax, o, 16));
      const float mn   = fmaxf(mrow[v], tmax);
      const float corr = __expf(mrow[v] - mn);
      const float p0   = __expf(s0[v] - mn);
      const float p1   = __expf(s1[v] - mn);
      float rs = p0 + p1;
      for (int o = 1; o < 16; o <<= 1) rs += __shfl_xor(rs, o, 16);
      lrow[v] = lrow[v] * corr + rs;
      mrow[v] = mn;
#pragma unroll
      for (int t = 0; t < 8; ++t) acc[t][v] *= corr;
      s0[v] = p0; s1[v] = p1;
    }

    // ---- C-layout -> A-layout re-tile of P through LDS (bf16) ----
    if (wave == 0) {
#pragma unroll
      for (int v = 0; v < 8; ++v) {
        const int r = v + 8 * half;
        pbuf[r][l16]      = bf16bits(s0[v]);
        pbuf[r][16 + l16] = bf16bits(s1[v]);
      }
    }
    __syncthreads();
    BF16x16 pa;
#pragma unroll
    for (int v = 0; v < 8; ++v)
      pa.u[v] = *(const uint32_t*)&pbuf[l16][kbase(v, half)];

    // ---- acc += P @ V' over this wave's D slice ----
    // Pair-interleaved V': dword (k,dcol),(k+1,dcol) is one aligned load.
#pragma unroll
    for (int t = 0; t < 8; ++t) {
      BF16x16 bv;
      const int dcol = wave * 128 + t * 16 + l16;
#pragma unroll
      for (int v = 0; v < 8; ++v) {
        const size_t pairrow = (rowB + keybase + kbase(v, half)) >> 1;
        bv.u[v] = VpP[pairrow * DD + dcol];
      }
      acc[t] = __builtin_amdgcn_wmma_f32_16x16x32_bf16(false, pa.v, false, bv.v,
                                                       (short)0, acc[t], false, false);
    }
  }

  // ---- epilogue: x = V + attn@V', then LayerNorm over D=1024 ----
#pragma unroll
  for (int v = 0; v < 8; ++v) {
    const int r = q0 + v + 8 * half;
    const float invl = 1.0f / lrow[v];
    float su = 0.0f, sq = 0.0f;
#pragma unroll
    for (int t = 0; t < 8; ++t) {
      const int col = wave * 128 + t * 16 + l16;
      float x = acc[t][v] * invl + Vres[(rowB + r) * DD + col];
      acc[t][v] = x;
      su += x; sq += x * x;
    }
    for (int o = 1; o < 16; o <<= 1) { su += __shfl_xor(su, o, 16); sq += __shfl_xor(sq, o, 16); }
    if (l16 == 0) { lnsum[wave][v + 8 * half] = su; lnsq[wave][v + 8 * half] = sq; }
  }
  __syncthreads();
#pragma unroll
  for (int v = 0; v < 8; ++v) {
    const int rl = v + 8 * half;
    const int r  = q0 + rl;
    float su = 0.0f, sq = 0.0f;
#pragma unroll
    for (int w = 0; w < 8; ++w) { su += lnsum[w][rl]; sq += lnsq[w][rl]; }
    const float mean = su * (1.0f / DD);
    const float var  = sq * (1.0f / DD) - mean * mean;
    const float rstd = rsqrtf(var + 1e-5f);
#pragma unroll
    for (int t = 0; t < 8; ++t) {
      const int col = wave * 128 + t * 16 + l16;
      const float y = (acc[t][v] - mean) * rstd * gamma[col] + beta[col];
      Out[(rowB + r) * DD + col] = y;
    }
  }
}

// ---------------------------------------------------------------------------
extern "C" void kernel_launch(void* const* d_in, const int* in_sizes, int n_in,
                              void* d_out, int out_size, void* d_ws, size_t ws_size,
                              hipStream_t stream) {
  const float*   Q     = (const float*)d_in[0];
  const float*   K     = (const float*)d_in[1];
  const float*   V     = (const float*)d_in[2];
  const uint8_t* mask  = (const uint8_t*)d_in[3];
  const float*   Wq    = (const float*)d_in[4];
  const float*   Wk    = (const float*)d_in[5];
  const float*   Wv    = (const float*)d_in[6];
  const float*   gamma = (const float*)d_in[7];
  const float*   beta  = (const float*)d_in[8];
  float*         out   = (float*)d_out;

  const size_t elems = (size_t)BB * SS * DD;   // 33,554,432
  uint16_t* Qp = (uint16_t*)d_ws;
  uint16_t* Kp = Qp + elems;
  uint16_t* Vp = Kp + elems;                   // pair-interleaved bf16

  const float inv_temp = 1.0f / (32.0f + 1e-6f);   // 1/(sqrt(1024)+eps)

  dim3 g1(BB * SS / 128, DD / 128);   // (256, 8)
  proj_gemm<<<g1, 256, 0, stream>>>(Q, Wq, Qp, inv_temp, 0); // temp folded into Q'
  proj_gemm<<<g1, 256, 0, stream>>>(K, Wk, Kp, 1.0f, 0);
  proj_gemm<<<g1, 256, 0, stream>>>(V, Wv, Vp, 1.0f, 1);     // pair-interleaved

  attn_ln<<<BB * SS / 16, 256, 0, stream>>>(Qp, Kp, (const uint32_t*)Vp, V,
                                            mask, gamma, beta, out);
}
